// DyFAIP_Aware_5634997092965
// MI455X (gfx1250) — compile-verified, matrix operand order checked by reading, not run on hardware
//
#include <hip/hip_runtime.h>
#include <math.h>

#define F_IN   32
#define HID    128
#define SEQLEN 64
#define NBATCH 64

typedef __attribute__((ext_vector_type(16))) __bf16 v16bf;
typedef __attribute__((ext_vector_type(8)))  __bf16 v8bf;
typedef __attribute__((ext_vector_type(8)))  float  v8f;

union Frag16 { v16bf v; v8bf h[2]; };

__device__ __forceinline__ unsigned short f2bf(float x) {
  union { float f; unsigned u; } c; c.f = x;
  unsigned r = c.u + 0x7fffu + ((c.u >> 16) & 1u);   // round-to-nearest-even
  return (unsigned short)(r >> 16);
}
__device__ __forceinline__ float sigm(float x) { return 1.f / (1.f + expf(-x)); }
__device__ __forceinline__ float eluf(float x) { return x > 0.f ? x : (expf(x) - 1.f); }

// 16-bit A/B fragment per CDNA5 ISA layout: lane L holds row (L&15),
// K = kt*32 + (L>=16 ? 8 : 0) .. +8 in v[0..3], +16 more in v[4..7].
__device__ __forceinline__ v16bf load_frag(const __bf16* base, int row, int ko) {
  Frag16 fr;
  fr.h[0] = *(const v8bf*)(base + row * HID + ko);
  fr.h[1] = *(const v8bf*)(base + row * HID + ko + 16);
  return fr.v;
}

// [64x128] (A in LDS, bf16 row-major) x [128x128] (B = W^T in global, bf16
// row-major over N) -> 4 m-tile accumulators for this wave's 16-wide N slab.
__device__ __forceinline__ void gemm_acc(const __bf16* A, const __bf16* B,
                                         int l15, int half, v8f acc[4]) {
#pragma unroll
  for (int kt = 0; kt < 4; ++kt) {
    const int ko = kt * 32 + half * 8;
    v16bf bf = load_frag(B, l15, ko);
#pragma unroll
    for (int mt = 0; mt < 4; ++mt) {
      v16bf af = load_frag(A, mt * 16 + l15, ko);
      acc[mt] = __builtin_amdgcn_wmma_f32_16x16x32_bf16(
          false, af, false, bf, (short)0, acc[mt], false, false);
    }
  }
}

// C/D 16x16 f32 layout: n = lane&15 (+tile), m = r + 8*(lane>=16) (+tile)
#define EPILOGUE(ACC, ...)                                                     \
  { _Pragma("unroll") for (int mt = 0; mt < 4; ++mt) {                         \
      _Pragma("unroll") for (int r = 0; r < 8; ++r) {                          \
        const int m = mt * 16 + r + half * 8;                                  \
        const int n = nb + l15;                                                \
        const int idx = m * HID + n;                                           \
        const float accv = ACC[mt][r];                                         \
        (void)idx; (void)n; (void)m; (void)accv;                               \
        __VA_ARGS__;                                                           \
      } } }

struct VecPtrs { const float* p[26]; };
enum { VDW = 0, VBD, VUJ, VUI, VUF, VUO, VUC, VULAST, VUTIME, VBJ, VBI, VBO,
       VBC, VBLAST, VBTIME, VBJM, VBJL, VWCI, VWCF, VWCO, VHTM, VCTM, VHTL,
       VCTL, VFA1, VFA2 };

constexpr size_t SMEM_MAIN = (size_t)(7 * 8192) * 4   // 7 f32 [64][128] buffers
                           + (size_t)(3 * 8192) * 2   // 3 bf16 [64][128] buffers
                           + (size_t)(26 * 128) * 4   // small per-feature vectors
                           + (size_t)(8 * 64) * 4;    // per-batch scalars
constexpr size_t SMEM_HEAD = (size_t)(2048 + 16384 + 512) * 4;

// ---------------------------------------------------------------------------
// Prep 1: freq-derived scalars. coef[b,f] and Sfw[b]. Single block.
// ---------------------------------------------------------------------------
__global__ __launch_bounds__(256) void dyfaip_prep_scalars(
    const float* __restrict__ freqs, float* __restrict__ coef,
    float* __restrict__ Sfw) {
  __shared__ float rmn[256], rmx[256], rfr[256];
  const int tid = threadIdx.x;
  float mn = 1e30f, mx = -1e30f, mfr = -1e30f;
  for (int i = tid; i < NBATCH * F_IN; i += 256) {
    float fv = freqs[i];
    int fidx = i & (F_IN - 1);
    float fr = ((float)SEQLEN - fv) * expf(-0.5f * (float)fidx);
    mn = fminf(mn, fv); mx = fmaxf(mx, fv); mfr = fmaxf(mfr, fr);
  }
  rmn[tid] = mn; rmx[tid] = mx; rfr[tid] = mfr;
  __syncthreads();
  for (int off = 128; off > 0; off >>= 1) {
    if (tid < off) {
      rmn[tid] = fminf(rmn[tid], rmn[tid + off]);
      rmx[tid] = fmaxf(rmx[tid], rmx[tid + off]);
      rfr[tid] = fmaxf(rfr[tid], rfr[tid + off]);
    }
    __syncthreads();
  }
  const float e1max = expf(-0.05f * rmn[0]);   // max of exp(-0.05*freq)
  const float e2max = expf(0.5f * rmx[0]);     // max of exp(+0.5*freq)
  const float frmax = rfr[0];
  for (int i = tid; i < NBATCH * F_IN; i += 256) {
    float fv = freqs[i];
    int fidx = i & (F_IN - 1);
    float factor_imp = expf(0.5f * fv) / e2max;          // thr = 0.93 * 1.0
    float frn = ((float)SEQLEN - fv) * expf(-0.5f * (float)fidx) / frmax;
    float ffeat = expf(-0.05f * fv) / e1max;
    coef[i] = (factor_imp >= 0.93f) ? frn : ffeat;
  }
  if (tid < NBATCH) {
    float s = 0.f;
    for (int ff = 0; ff < F_IN; ++ff) s += expf(-0.5f * freqs[tid * F_IN + ff]);
    Sfw[tid] = s;
  }
}

// ---------------------------------------------------------------------------
// Prep 2: repack one weight [F][H][K] f32 -> Wt[F][K=n][H] bf16 (B^T layout)
// ---------------------------------------------------------------------------
__global__ __launch_bounds__(256) void dyfaip_repack(
    const float* __restrict__ W, unsigned short* __restrict__ Wt) {
  const int f = blockIdx.x;
  const float* src = W + (size_t)f * HID * HID;
  unsigned short* dst = Wt + (size_t)f * HID * HID;
  for (int i = threadIdx.x; i < HID * HID; i += 256) {
    int n = i >> 7, h = i & 127;
    dst[n * HID + h] = f2bf(src[h * HID + n]);
  }
}

// ---------------------------------------------------------------------------
// Main: persistent recurrent kernel. One workgroup per feature f.
// ---------------------------------------------------------------------------
__global__ __launch_bounds__(256) void dyfaip_recurrent(
    const float* __restrict__ inputs, const float* __restrict__ times,
    const float* __restrict__ lastv, VecPtrs vp,
    const __bf16* __restrict__ WtJ, const __bf16* __restrict__ WtI,
    const __bf16* __restrict__ WtF, const __bf16* __restrict__ WtO,
    const __bf16* __restrict__ WtC, const __bf16* __restrict__ WtD,
    const float* __restrict__ F_alpha_nb, const float* __restrict__ coefG,
    const float* __restrict__ SfwG, float* __restrict__ alphaT) {
  const int f = blockIdx.x;
  const int tid = threadIdx.x;
  const int wv = tid >> 5;
  const int lane = tid & 31;
  const int half = lane >> 4;
  const int l15 = lane & 15;
  const int nb = wv * 16;             // this wave's N slab

  extern __shared__ char smem[];
  float* sH   = (float*)smem;         // h state (f32)
  float* sC   = sH  + 8192;           // c state -> ct
  float* sT0  = sC  + 8192;           // dh -> dj
  float* sT1  = sT0 + 8192;           // dc -> f_new
  float* sLT  = sT1 + 8192;           // last_tilda
  float* sI   = sLT + 8192;           // i gate
  float* sJ32 = sI  + 8192;           // j (f32)
  unsigned short* sH16 = (unsigned short*)(sJ32 + 8192);  // bf16 copies (A ops)
  unsigned short* sC16 = sH16 + 8192;
  unsigned short* sJ16 = sC16 + 8192;
  float* sVec = (float*)(sJ16 + 8192);      // 26 x 128 feature vectors
  float* sSfw   = sVec + 26 * HID;          // per-batch scalars (64 each)
  float* sCoef  = sSfw + 64;
  float* sSlast = sCoef + 64;
  float* sSimp  = sSlast + 64;
  float* sMT    = sSimp + 64;
  float* sSMT   = sMT + 64;
  float* sXb    = sSMT + 64;
  float* sLb    = sXb + 64;

  const float* sDw    = sVec + VDW * HID;
  const float* sBd    = sVec + VBD * HID;
  const float* sUJ    = sVec + VUJ * HID;
  const float* sUI    = sVec + VUI * HID;
  const float* sUF    = sVec + VUF * HID;
  const float* sUO    = sVec + VUO * HID;
  const float* sUC    = sVec + VUC * HID;
  const float* sULAST = sVec + VULAST * HID;
  const float* sUT    = sVec + VUTIME * HID;
  const float* sBJ    = sVec + VBJ * HID;
  const float* sBI    = sVec + VBI * HID;
  const float* sBO    = sVec + VBO * HID;
  const float* sBC    = sVec + VBC * HID;
  const float* sBLAST = sVec + VBLAST * HID;
  const float* sBT    = sVec + VBTIME * HID;
  const float* sBJM   = sVec + VBJM * HID;
  const float* sBJL   = sVec + VBJL * HID;
  const float* sWCI   = sVec + VWCI * HID;
  const float* sWCF   = sVec + VWCF * HID;
  const float* sWCO   = sVec + VWCO * HID;
  const float* sHTM   = sVec + VHTM * HID;
  const float* sCTM   = sVec + VCTM * HID;
  const float* sHTL   = sVec + VHTL * HID;
  const float* sCTL   = sVec + VCTL * HID;
  const float* sFA1   = sVec + VFA1 * HID;
  const float* sFA2   = sVec + VFA2 * HID;

  // ---- init: zero state, stage feature vectors -----------------------------
  for (int i = tid; i < 8192; i += 256) {
    sH[i] = 0.f; sC[i] = 0.f; sH16[i] = 0; sC16[i] = 0;
  }
  for (int v = 0; v < 26; ++v) {
    const float* src;
    if (v == VFA1)      src = vp.p[VFA1] + f * 256;
    else if (v == VFA2) src = vp.p[VFA2] + f * 256 + HID;
    else                src = vp.p[v] + f * HID;
    for (int k = tid; k < HID; k += 256) sVec[v * HID + k] = src[k];
  }
  if (tid < NBATCH) { sSfw[tid] = SfwG[tid]; sCoef[tid] = coefG[tid * F_IN + f]; }

  const size_t wstride = (size_t)HID * HID;
  const __bf16* BWj = WtJ + (size_t)f * wstride + (size_t)nb * HID;
  const __bf16* BWi = WtI + (size_t)f * wstride + (size_t)nb * HID;
  const __bf16* BWf = WtF + (size_t)f * wstride + (size_t)nb * HID;
  const __bf16* BWo = WtO + (size_t)f * wstride + (size_t)nb * HID;
  const __bf16* BWc = WtC + (size_t)f * wstride + (size_t)nb * HID;
  const __bf16* BWd = WtD + (size_t)f * wstride + (size_t)nb * HID;
  const float fanb = F_alpha_nb[f];

  for (int t = 0; t < SEQLEN; ++t) {
    if (tid < NBATCH) {
      const int b = tid;
      float tt = times[b * SEQLEN + t];
      float mT = 1.f / logf(tt + 2.718281828459045f);
      sMT[b] = mT; sSMT[b] = sigm(mT);
      sXb[b] = inputs[(b * SEQLEN + t) * F_IN + f];
      sLb[b] = lastv[(b * SEQLEN + t) * F_IN + f];
    }
    __syncthreads();

    // ---- phase 1: dh = sig(t1 + h.Wd + b_d), dc = sig(t1 + c.Wd + b_d) ----
    {
      v8f a0[4] = {}; v8f a1[4] = {};
#pragma unroll
      for (int kt = 0; kt < 4; ++kt) {
        const int ko = kt * 32 + half * 8;
        v16bf bf = load_frag(BWd, l15, ko);          // share Wd fragments
#pragma unroll
        for (int mt = 0; mt < 4; ++mt) {
          v16bf ah = load_frag((const __bf16*)sH16, mt * 16 + l15, ko);
          v16bf ac = load_frag((const __bf16*)sC16, mt * 16 + l15, ko);
          a0[mt] = __builtin_amdgcn_wmma_f32_16x16x32_bf16(
              false, ah, false, bf, (short)0, a0[mt], false, false);
          a1[mt] = __builtin_amdgcn_wmma_f32_16x16x32_bf16(
              false, ac, false, bf, (short)0, a1[mt], false, false);
        }
      }
#pragma unroll
      for (int mt = 0; mt < 4; ++mt)
#pragma unroll
        for (int r = 0; r < 8; ++r) {
          const int m = mt * 16 + r + half * 8;
          const int n = nb + l15;
          float t1v = sSfw[m] * sDw[n] + sBd[n];
          sT0[m * HID + n] = sigm(a0[mt][r] + t1v);
          sT1[m * HID + n] = sigm(a1[mt][r] + t1v);
        }
    }
    __syncthreads();

    // ---- phase 2: per-batch rank-1 reductions (S_last, S_imp) -------------
    if (tid < NBATCH) {
      const int b = tid;
      const float* dh = sT0 + b * HID;
      const float* dc = sT1 + b * HID;
      float sl = 0.f, sm = 0.f;
      for (int k = 0; k < HID; ++k) {
        sl += dh[k] * sHTL[k] + dc[k] * sCTL[k];
        sm += dh[k] * sHTM[k] + dc[k] * sCTM[k];
      }
      const float lb = sLb[b], xb = sXb[b];
      float Sl, Si;
      if (lb != lb) {  // NaN -> imputed
        float s = 0.f;
        for (int k = 0; k < HID; ++k) s += tanhf(sl + sBJL[k]);
        Sl = sCoef[b] * s;
      } else Sl = (float)HID * lb;
      if (xb != xb) {
        float s = 0.f;
        for (int k = 0; k < HID; ++k) s += tanhf(sm + sBJM[k]);
        Si = sCoef[b] * s;
      } else Si = (float)HID * xb;
      sSlast[b] = Sl; sSimp[b] = Si;
    }
    __syncthreads();

    // ---- phase 3: last_tilda, h += last_tilda, refresh h bf16 -------------
    for (int i = tid; i < 8192; i += 256) {
      const int b = i >> 7, k = i & 127;
      float lt = eluf(sSlast[b] * sULAST[k] + sBLAST[k]);
      float h = sH[i] + lt;
      sLT[i] = lt; sH[i] = h; sH16[i] = f2bf(h);
    }
    __syncthreads();

    // ---- j = tanh(h.Wj + S_imp*U_j + b_j) ---------------------------------
    {
      v8f acc[4] = {};
      gemm_acc((const __bf16*)sH16, BWj, l15, half, acc);
      EPILOGUE(acc, {
        float jv = tanhf(accv + sSimp[m] * sUJ[n] + sBJ[n]);
        sJ32[idx] = jv; sJ16[idx] = f2bf(jv);
      })
    }
    __syncthreads();   // dj GEMM reads all of sJ16

    // ---- dj = sig(t1 + j.Wd + b_d) ----------------------------------------
    {
      v8f acc[4] = {};
      gemm_acc((const __bf16*)sJ16, BWd, l15, half, acc);
      EPILOGUE(acc, { sT0[idx] = sigm(accv + sSfw[m] * sDw[n] + sBd[n]); })
    }
    // ---- i gate -----------------------------------------------------------
    {
      v8f acc[4] = {};
      gemm_acc((const __bf16*)sH16, BWi, l15, half, acc);
      EPILOGUE(acc, {
        sI[idx] = sigm(accv + sSimp[m] * sUI[n] + sC[idx] * sWCI[n] +
                       sBI[n] * sT0[idx]);
      })
    }
    // ---- f gate -> f_new --------------------------------------------------
    {
      v8f acc[4] = {};
      gemm_acc((const __bf16*)sH16, BWf, l15, half, acc);
      EPILOGUE(acc, {
        float fv = sigm(accv + sSimp[m] * sUF[n] + sC[idx] * sWCF[n]);
        sT1[idx] = fv * sMT[m] + (1.f - fv) * sT0[idx];
      })
    }
    // ---- cbar -> ct -------------------------------------------------------
    {
      v8f acc[4] = {};
      gemm_acc((const __bf16*)sH16, BWc, l15, half, acc);
      EPILOGUE(acc, {
        float cb = tanhf(accv + sSimp[m] * sUC[n] + sBC[n]);
        float tg = sigm(sSimp[m] * sUT[n] + sSMT[m] + sBT[n]);
        sC[idx] = (sT1[idx] + tg) * sC[idx] + sI[idx] * sJ32[idx] * tg * cb;
      })
    }
    // ---- o gate -> h_new --------------------------------------------------
    {
      v8f acc[4] = {};
      gemm_acc((const __bf16*)sH16, BWo, l15, half, acc);
      EPILOGUE(acc, {
        float ov = sigm(accv + sSimp[m] * sUO[n] + sC[idx] * sWCO[n] + sBO[n]);
        sH[idx] = ov * tanhf(sC[idx] + sLT[idx]);
      })
    }
    __syncthreads();

    // ---- alpha contribution + refresh bf16 state --------------------------
    if (tid < NBATCH) {
      const int b = tid;
      float s = fanb;
      const float* hh = sH + b * HID;
      const float* cc = sC + b * HID;
      for (int k = 0; k < HID; ++k) s += hh[k] * sFA1[k] + cc[k] * sFA2[k];
      alphaT[(b * F_IN + f) * SEQLEN + t] = tanhf(s);
    }
    for (int i = tid; i < 8192; i += 256) {
      sH16[i] = f2bf(sH[i]); sC16[i] = f2bf(sC[i]);
    }
    // top-of-loop __syncthreads() orders the refresh vs next phase 1
  }
}

// ---------------------------------------------------------------------------
// Head: per-batch LN1 -> beta matmul -> LN2 -> Phi. One block per batch.
// ---------------------------------------------------------------------------
__global__ __launch_bounds__(256) void dyfaip_head(
    const float* __restrict__ alphaT, const float* __restrict__ ln1w,
    const float* __restrict__ ln1b, const float* __restrict__ Fbw,
    const float* __restrict__ Fbb, const float* __restrict__ ln2w,
    const float* __restrict__ ln2b, const float* __restrict__ Phiw,
    const float* __restrict__ Phib, float* __restrict__ out) {
  extern __shared__ char smem[];
  float* sA = (float*)smem;       // [32][64]
  float* sMu = sA + 2048;         // [32][512]
  float* red = sMu + 16384;       // 2x256
  const int tid = threadIdx.x;
  const int b = blockIdx.x;

  float s = 0.f, ss = 0.f;
#pragma unroll
  for (int i = 0; i < 8; ++i) {
    const int idx = tid + i * 256;
    float v = alphaT[b * 2048 + idx];
    sA[idx] = v; s += v; ss += v * v;
  }
  red[tid] = s; red[256 + tid] = ss;
  __syncthreads();
  for (int off = 128; off > 0; off >>= 1) {
    if (tid < off) { red[tid] += red[tid + off]; red[256 + tid] += red[256 + tid + off]; }
    __syncthreads();
  }
  const float mean1 = red[0] * (1.f / 2048.f);
  const float var1 = red[256] * (1.f / 2048.f) - mean1 * mean1;
  const float rstd1 = rsqrtf(var1 + 1e-5f);
  __syncthreads();
#pragma unroll
  for (int i = 0; i < 8; ++i) {
    const int idx = tid + i * 256;
    const int ff = idx >> 6, tt = idx & 63;
    sA[idx] = (sA[idx] - mean1) * rstd1 * ln1w[ff * 64 + tt] + ln1b[ff * 64 + tt];
  }
  __syncthreads();

  s = 0.f; ss = 0.f;
  for (int o = tid; o < 16384; o += 256) {
    const int ff = o >> 9, q = o & 511;
    const float* arow = sA + ff * 64;
    const float* wrow = Fbw + q * 64;
    float acc = Fbb[q];
    for (int tt = 0; tt < 64; ++tt) acc += arow[tt] * wrow[tt];
    sMu[o] = acc; s += acc; ss += acc * acc;
  }
  red[tid] = s; red[256 + tid] = ss;
  __syncthreads();
  for (int off = 128; off > 0; off >>= 1) {
    if (tid < off) { red[tid] += red[tid + off]; red[256 + tid] += red[256 + tid + off]; }
    __syncthreads();
  }
  const float mean2 = red[0] * (1.f / 16384.f);
  const float var2 = red[256] * (1.f / 16384.f) - mean2 * mean2;
  const float rstd2 = rsqrtf(var2 + 1e-5f);
  __syncthreads();

  {
    const int ff = tid >> 3, part = tid & 7;
    float acc = 0.f;
    for (int q = part * 64; q < part * 64 + 64; ++q) {
      const int idx = ff * 512 + q;
      float mu = (sMu[idx] - mean2) * rstd2 * ln2w[idx] + ln2b[idx];
      acc += mu * Phiw[q];
    }
    red[tid] = acc;
  }
  __syncthreads();
  if (tid < F_IN) {
    float acc = Phib[0];
#pragma unroll
    for (int p2 = 0; p2 < 8; ++p2) acc += red[tid * 8 + p2];
    out[b * F_IN + tid] = acc;
  }
}

// ---------------------------------------------------------------------------
extern "C" void kernel_launch(void* const* d_in, const int* in_sizes, int n_in,
                              void* d_out, int out_size, void* d_ws, size_t ws_size,
                              hipStream_t stream) {
  (void)in_sizes; (void)n_in; (void)out_size; (void)ws_size;
  const float* inputs = (const float*)d_in[0];
  const float* times  = (const float*)d_in[1];
  const float* lastv  = (const float*)d_in[2];
  const float* freqs  = (const float*)d_in[3];

  VecPtrs vp;
  vp.p[VUJ]    = (const float*)d_in[4];
  vp.p[VUI]    = (const float*)d_in[5];
  vp.p[VUF]    = (const float*)d_in[6];
  vp.p[VUO]    = (const float*)d_in[7];
  vp.p[VUC]    = (const float*)d_in[8];
  vp.p[VULAST] = (const float*)d_in[9];
  vp.p[VUTIME] = (const float*)d_in[10];
  vp.p[VDW]    = (const float*)d_in[11];
  vp.p[VWCI]   = (const float*)d_in[18];
  vp.p[VWCF]   = (const float*)d_in[19];
  vp.p[VWCO]   = (const float*)d_in[20];
  vp.p[VBJ]    = (const float*)d_in[21];
  vp.p[VBI]    = (const float*)d_in[22];
  vp.p[VBO]    = (const float*)d_in[23];
  vp.p[VBC]    = (const float*)d_in[24];
  vp.p[VBLAST] = (const float*)d_in[25];
  vp.p[VBTIME] = (const float*)d_in[26];
  vp.p[VBD]    = (const float*)d_in[27];
  vp.p[VBJM]   = (const float*)d_in[28];
  vp.p[VBJL]   = (const float*)d_in[29];
  vp.p[VHTM]   = (const float*)d_in[30];
  vp.p[VCTM]   = (const float*)d_in[31];
  vp.p[VHTL]   = (const float*)d_in[32];
  vp.p[VCTL]   = (const float*)d_in[33];
  vp.p[VFA1]   = (const float*)d_in[34];
  vp.p[VFA2]   = (const float*)d_in[34];
  const float* F_alpha_nb = (const float*)d_in[35];

  char* ws = (char*)d_ws;
  const size_t WELEMS = (size_t)F_IN * HID * HID;
  unsigned short* Wt[6];  // order: j, i, f, o, c, d (matches d_in[12..17])
  for (int w = 0; w < 6; ++w) Wt[w] = (unsigned short*)(ws + (size_t)w * WELEMS * 2);
  float* alphaT = (float*)(ws + 6 * WELEMS * 2);
  float* coef = alphaT + (size_t)NBATCH * F_IN * SEQLEN;
  float* Sfw = coef + (size_t)NBATCH * F_IN;

  dyfaip_prep_scalars<<<1, 256, 0, stream>>>(freqs, coef, Sfw);
  for (int w = 0; w < 6; ++w)
    dyfaip_repack<<<F_IN, 256, 0, stream>>>((const float*)d_in[12 + w], Wt[w]);

  (void)hipFuncSetAttribute((const void*)dyfaip_recurrent,
                            hipFuncAttributeMaxDynamicSharedMemorySize,
                            (int)SMEM_MAIN);
  dyfaip_recurrent<<<F_IN, 256, SMEM_MAIN, stream>>>(
      inputs, times, lastv, vp,
      (const __bf16*)Wt[0], (const __bf16*)Wt[1], (const __bf16*)Wt[2],
      (const __bf16*)Wt[3], (const __bf16*)Wt[4], (const __bf16*)Wt[5],
      F_alpha_nb, coef, Sfw, alphaT);

  (void)hipFuncSetAttribute((const void*)dyfaip_head,
                            hipFuncAttributeMaxDynamicSharedMemorySize,
                            (int)SMEM_HEAD);
  dyfaip_head<<<NBATCH, 256, SMEM_HEAD, stream>>>(
      alphaT, (const float*)d_in[40], (const float*)d_in[41],
      (const float*)d_in[36], (const float*)d_in[37],
      (const float*)d_in[42], (const float*)d_in[43],
      (const float*)d_in[38], (const float*)d_in[39], (float*)d_out);
}
